// myBioCell_65670049956454
// MI455X (gfx1250) — matrix operand order repaired and checked by reading, not compiled, not run on hardware
//
#include <hip/hip_runtime.h>

typedef __attribute__((ext_vector_type(16))) __bf16 v16bf;
typedef __attribute__((ext_vector_type(8)))  __bf16 v8bf;
typedef __attribute__((ext_vector_type(4)))  __bf16 v4bf;
typedef __attribute__((ext_vector_type(8)))  float  v8f;
typedef __attribute__((ext_vector_type(4)))  float  v4f;

#define TOKENS 32768
#define D_IN   1024
#define D_OUT  1024

#define BM 256                 // workgroup token-rows
#define BN 128                 // workgroup output-cols
#define BK 64                  // K tile staged in LDS
#define LDS_STRIDE (BK + 8)    // 72 bf16 = 144 bytes: 16B-aligned rows + bank padding

__global__ __launch_bounds__(256)
void bio_linear_wmma_bf16(const float* __restrict__ X,   // [TOKENS, D_IN]
                          const float* __restrict__ W,   // [D_OUT, D_IN]
                          const float* __restrict__ R,   // [D_OUT]
                          float* __restrict__ Y)         // [TOKENS, D_OUT]
{
    __shared__ __align__(16) __bf16 sA[BM * LDS_STRIDE];   // 36 KB
    __shared__ __align__(16) __bf16 sB[BN * LDS_STRIDE];   // 18 KB

    const int tid  = threadIdx.x;
    const int lane = tid & 31;
    const int wave = tid >> 5;

    const int n0 = blockIdx.x * BN;   // output-column tile
    const int m0 = blockIdx.y * BM;   // token-row tile

    const int wm = wave & 3;          // 4 slabs of 64 rows
    const int wn = wave >> 2;         // 2 slabs of 64 cols

    const int half = lane >> 4;       // which 16-lane group
    const int l16  = lane & 15;

    v8f acc[4][4] = {};               // 64x64 per wave: 4 m-tiles x 4 n-tiles

    for (int kb = 0; kb < D_IN; kb += BK) {
        __syncthreads();   // LDS WAR guard vs previous iteration's fragment reads

        // ---- global fp32 -> LDS bf16 staging ----
        // A tile: 256 rows x 64 k  (4096 float4 slots, 16 per thread)
        #pragma unroll
        for (int it = 0; it < 16; ++it) {
            int idx = tid + it * 256;
            int row = idx >> 4;                // 16 float4 per row
            int c4  = idx & 15;
            v4f f = *(const v4f*)(X + (size_t)(m0 + row) * D_IN + kb + c4 * 4);
            v4bf h = { (__bf16)f[0], (__bf16)f[1], (__bf16)f[2], (__bf16)f[3] };
            *(v4bf*)(sA + row * LDS_STRIDE + c4 * 4) = h;
        }
        // B tile: 128 rows x 64 k  (2048 float4 slots, 8 per thread)
        #pragma unroll
        for (int it = 0; it < 8; ++it) {
            int idx = tid + it * 256;
            int row = idx >> 4;
            int c4  = idx & 15;
            v4f f = *(const v4f*)(W + (size_t)(n0 + row) * D_IN + kb + c4 * 4);
            v4bf h = { (__bf16)f[0], (__bf16)f[1], (__bf16)f[2], (__bf16)f[3] };
            *(v4bf*)(sB + row * LDS_STRIDE + c4 * 4) = h;
        }
        __syncthreads();

        // prefetch next K tile into cache while WMMAs run (global_prefetch_b8)
        if (kb + BK < D_IN) {
            __builtin_prefetch(X + (size_t)(m0 + tid) * D_IN + kb + BK, 0, 3);
            if (tid < BN)
                __builtin_prefetch(W + (size_t)(n0 + tid) * D_IN + kb + BK, 0, 3);
        }

        // ---- two K=32 WMMA steps per LDS tile ----
        #pragma unroll
        for (int kk = 0; kk < BK; kk += 32) {
            // A fragments (16-bit A layout): lane = row l16,
            // K chunks [kk+8h, +8) and [kk+16+8h, +8)
            v16bf afrag[4];
            #pragma unroll
            for (int mt = 0; mt < 4; ++mt) {
                const __bf16* base = sA + (wm * 64 + mt * 16 + l16) * LDS_STRIDE;
                union { v16bf v; v8bf h[2]; } u;
                u.h[0] = *(const v8bf*)(base + kk + 8 * half);
                u.h[1] = *(const v8bf*)(base + kk + 16 + 8 * half);
                afrag[mt] = u.v;
            }
            // B fragments (32x16 B layout): lane = column l16 (= weight row),
            // K chunks [kk+8h, +8) and [kk+16+8h, +8)
            v16bf bfrag[4];
            #pragma unroll
            for (int nt = 0; nt < 4; ++nt) {
                const __bf16* base = sB + (wn * 64 + nt * 16 + l16) * LDS_STRIDE;
                union { v16bf v; v8bf h[2]; } u;
                u.h[0] = *(const v8bf*)(base + kk + 8 * half);
                u.h[1] = *(const v8bf*)(base + kk + 16 + 8 * half);
                bfrag[nt] = u.v;
            }
            // 16 independent WMMA chains — 1.0 ds_load_b128 per WMMA
            #pragma unroll
            for (int mt = 0; mt < 4; ++mt)
                #pragma unroll
                for (int nt = 0; nt < 4; ++nt)
                    acc[mt][nt] = __builtin_amdgcn_wmma_f32_16x16x32_bf16(
                        false, afrag[mt], false, bfrag[nt],
                        (short)0, acc[mt][nt], false, false);
        }
    }

    // ---- epilogue: 'remain' threshold + store ----
    // C/D layout: VGPR v, lanes 0-15 -> row v, col lane; lanes 16-31 -> row v+8
    float rn[4];
    #pragma unroll
    for (int nt = 0; nt < 4; ++nt)
        rn[nt] = R[n0 + wn * 64 + nt * 16 + l16];

    #pragma unroll
    for (int mt = 0; mt < 4; ++mt) {
        #pragma unroll
        for (int nt = 0; nt < 4; ++nt) {
            const int col = n0 + wn * 64 + nt * 16 + l16;
            #pragma unroll
            for (int v = 0; v < 8; ++v) {
                const int row = m0 + wm * 64 + mt * 16 + v + 8 * half;
                const float val = acc[mt][nt][v];
                Y[(size_t)row * D_OUT + col] = (val > rn[nt]) ? val : 0.0f;
            }
        }
    }
}

extern "C" void kernel_launch(void* const* d_in, const int* in_sizes, int n_in,
                              void* d_out, int out_size, void* d_ws, size_t ws_size,
                              hipStream_t stream) {
    (void)in_sizes; (void)n_in; (void)d_ws; (void)ws_size; (void)out_size;
    const float* X = (const float*)d_in[0];   // [32768, 1024] fp32
    const float* W = (const float*)d_in[1];   // [1024, 1024]  fp32
    const float* R = (const float*)d_in[2];   // [1024]        fp32
    float*       Y = (float*)d_out;           // [32768, 1024] fp32

    dim3 grid(D_OUT / BN, TOKENS / BM);       // 8 x 128
    dim3 block(256);
    bio_linear_wmma_bf16<<<grid, block, 0, stream>>>(X, W, R, Y);
}
